// PairBiasAttention_59176059405031
// MI455X (gfx1250) — matrix operand order repaired and checked by reading, not compiled, not run on hardware
//
#include <hip/hip_runtime.h>
#include <hip/hip_bf16.h>

// ---------------------------------------------------------------------------
// PairBiasAttention for MI455X (gfx1250, wave32, WMMA 16x16x32 f16).
// z stream (512 MB) dominates HBM traffic -> double-buffered async-to-LDS.
// GEMMs/attention run on the WMMA pipe.
// ---------------------------------------------------------------------------

typedef __attribute__((ext_vector_type(16))) _Float16 v16h;
typedef __attribute__((ext_vector_type(8)))  _Float16 v8h;
typedef __attribute__((ext_vector_type(8)))  float    v8f;
typedef __attribute__((ext_vector_type(4)))  float    v4f;

__device__ __forceinline__ v8f zero8() {
  v8f z;
#pragma unroll
  for (int i = 0; i < 8; ++i) z[i] = 0.0f;
  return z;
}

// A-fragment (16x32 f16, row-major source). Lane L: row m = L%16 (caller bakes
// into Arow), K-halves: group g = L/16 -> K = {k0+8g .. k0+8g+7, k0+16+8g .. +7}.
__device__ __forceinline__ v16h load_afrag(const _Float16* __restrict__ Arow, int k0, int grp) {
  v8h lo = *(const v8h*)(Arow + k0 + grp * 8);
  v8h hi = *(const v8h*)(Arow + k0 + 16 + grp * 8);
  v16h a;
#pragma unroll
  for (int j = 0; j < 8; ++j) { a[j] = lo[j]; a[8 + j] = hi[j]; }
  return a;
}

// CDNA5 async memory -> LDS (ASYNCcnt-tracked). 16 bytes per lane.
__device__ __forceinline__ void async_load_b128(unsigned lds_off, const void* gaddr) {
  asm volatile("global_load_async_to_lds_b128 %0, %1, off"
               :: "v"(lds_off), "v"(gaddr)
               : "memory");
}
__device__ __forceinline__ void wait_async_le1() {
  asm volatile("s_wait_asynccnt 0x1" ::: "memory");
}
__device__ __forceinline__ void wait_async_0() {
  asm volatile("s_wait_asynccnt 0x0" ::: "memory");
}

// ---------------------------------------------------------------------------
// fp32 -> fp16 conversion (for weights)
// ---------------------------------------------------------------------------
__global__ void cvt_f32_f16_kernel(const float* __restrict__ src, _Float16* __restrict__ dst, int n) {
  int i = blockIdx.x * blockDim.x + threadIdx.x;
  if (i < n) dst[i] = (_Float16)src[i];
}

// ---------------------------------------------------------------------------
// LayerNorm of s (C=768), one wave per row. Output fp16.
// ---------------------------------------------------------------------------
__global__ void ln_s_kernel(const float* __restrict__ s, const float* __restrict__ w,
                            const float* __restrict__ b, _Float16* __restrict__ snh) {
  const int C = 768;
  int wid = threadIdx.x >> 5, lane = threadIdx.x & 31;
  int row = blockIdx.x * 8 + wid;
  const float* x = s + (size_t)row * C;
  float vals[24];
  float sum = 0.f, sq = 0.f;
#pragma unroll
  for (int j = 0; j < 24; ++j) {
    float v = x[j * 32 + lane];
    vals[j] = v; sum += v; sq += v * v;
  }
#pragma unroll
  for (int off = 16; off >= 1; off >>= 1) {
    sum += __shfl_xor(sum, off, 32);
    sq  += __shfl_xor(sq,  off, 32);
  }
  float mu = sum * (1.0f / C);
  float var = sq * (1.0f / C) - mu * mu;
  float rs = rsqrtf(var + 1e-5f);
  _Float16* o = snh + (size_t)row * C;
#pragma unroll
  for (int j = 0; j < 24; ++j) {
    int c = j * 32 + lane;
    o[c] = (_Float16)((vals[j] - mu) * rs * w[c] + b[c]);
  }
}

// ---------------------------------------------------------------------------
// WMMA GEMM: C[MxN] = A[MxK](f16) @ B[KxN](f16) (+bias).
// One wave computes a 16(M) x 64(N) strip: 4 accumulators share one A fragment.
// K multiple of 32, M multiple of 16, N multiple of 64.
// mode 0: store fp16; mode 1: store fp32 sigmoid(x); mode 2: store fp32 x.
// ---------------------------------------------------------------------------
__global__ void gemm_wmma_kernel(const _Float16* __restrict__ A, int lda,
                                 const _Float16* __restrict__ Bm, int ldb,
                                 int M, int Nn, int K,
                                 const float* __restrict__ bias,
                                 _Float16* __restrict__ outh, float* __restrict__ outf,
                                 int ldc, int mode) {
  int wid = threadIdx.x >> 5, lane = threadIdx.x & 31;
  int n16 = lane & 15, grp = lane >> 4;
  int groupsN = Nn / 64;
  int wg = blockIdx.x * 4 + wid;
  if (wg >= groupsN * (M / 16)) return;
  int tm = wg / groupsN, tg = wg % groupsN;

  const _Float16* Arow = A + (size_t)(tm * 16 + n16) * lda;
  const _Float16* Bp = Bm + tg * 64;

  v8f acc[4];
#pragma unroll
  for (int t = 0; t < 4; ++t) acc[t] = zero8();

  for (int k0 = 0; k0 < K; k0 += 32) {
    v16h a = load_afrag(Arow, k0, grp);
    const _Float16* bcol = Bp + n16;
#pragma unroll
    for (int t = 0; t < 4; ++t) {
      v16h bfr;
#pragma unroll
      for (int s = 0; s < 16; ++s) bfr[s] = bcol[(size_t)(k0 + (s & 7) + 8 * grp + 16 * (s >> 3)) * ldb + t * 16];
      acc[t] = __builtin_amdgcn_wmma_f32_16x16x32_f16(false, a, false, bfr, (short)0, acc[t], false, false);
    }
  }

#pragma unroll
  for (int t = 0; t < 4; ++t) {
    int n = tg * 64 + t * 16 + n16;
    float bv = bias ? bias[n] : 0.0f;
#pragma unroll
    for (int r = 0; r < 8; ++r) {
      int m = tm * 16 + r + 8 * grp;
      float v = acc[t][r] + bv;
      size_t idx = (size_t)m * ldc + n;
      if (mode == 0)      outh[idx] = (_Float16)v;
      else if (mode == 1) outf[idx] = 1.0f / (1.0f + __expf(-v));
      else                outf[idx] = v;
    }
  }
}

// ---------------------------------------------------------------------------
// K^T repack: kT[h][d][n] (d padded 48->64 with zeros) so attention B-fragments
// read consecutive keys n across lanes for each d (elements N apart per lane).
// ---------------------------------------------------------------------------
__global__ void ktrans_kernel(const _Float16* __restrict__ qkvh, _Float16* __restrict__ kT) {
  int tid = blockIdx.x * blockDim.x + threadIdx.x;   // h(16) * d(64) * n(1024)
  int n = tid & 1023;
  int d = (tid >> 10) & 63;
  int h = tid >> 16;
  _Float16 v = (_Float16)0.0f;
  if (d < 48) v = qkvh[(size_t)n * 2304 + 768 + h * 48 + d];
  kT[tid] = v;
}

// ---------------------------------------------------------------------------
// Pair bias: bias[h][q][k] = (LayerNorm(z[q,k,:]) @ W_zb)[h], masked.
// One wave per 4 consecutive (q,k) pairs. The 512B z rows are double-buffered
// into LDS with CDNA5 async loads (ASYNCcnt), plus global_prefetch one row ahead.
// ---------------------------------------------------------------------------
__global__ void zbias_kernel(const float* __restrict__ z, const float* __restrict__ w,
                             const float* __restrict__ b, const float* __restrict__ Wzb,
                             const unsigned char* __restrict__ mask,
                             _Float16* __restrict__ biash) {
  const int CZ = 128, H = 16, N = 1024;
  __shared__ float wzb_s[CZ * H];
  __shared__ float lw[CZ], lb[CZ];
  __shared__ float zbuf[8][2][CZ];     // 8 waves x 2 ring slots x 512B
  for (int i = threadIdx.x; i < CZ * H; i += blockDim.x) wzb_s[i] = Wzb[i];
  if (threadIdx.x < CZ) { lw[threadIdx.x] = w[threadIdx.x]; lb[threadIdx.x] = b[threadIdx.x]; }

  int wid = threadIdx.x >> 5, lane = threadIdx.x & 31;
  int c0 = lane * 4;
  size_t pair0 = ((size_t)blockIdx.x * 8 + wid) * 4;

  // LDS byte offsets for this lane's 16B slice of each ring slot
  // (flat pointer low 32 bits == DS address on CDNA5)
  unsigned lds0 = (unsigned)(unsigned long long)(const void*)&zbuf[wid][0][c0];
  unsigned lds1 = (unsigned)(unsigned long long)(const void*)&zbuf[wid][1][c0];

  // prologue: kick off row 0
  async_load_b128(lds0, z + pair0 * CZ + c0);
  __syncthreads();   // wzb_s / lw / lb ready

  for (int it = 0; it < 4; ++it) {
    size_t pair = pair0 + it;
    const float* xnext = z + (pair + 1) * CZ + c0;
    if (it < 3) {
      async_load_b128((it & 1) ? lds0 : lds1, xnext);     // slot for row it+1
      __builtin_prefetch(xnext + CZ, 0, 1);               // row it+2 -> global_prefetch
      wait_async_le1();                                   // in-order: row it resident
    } else {
      wait_async_0();
    }

    v4f xv4 = *(const v4f*)&zbuf[wid][it & 1][c0];
    float xv[4];
#pragma unroll
    for (int j = 0; j < 4; ++j) xv[j] = xv4[j];

    float sum = 0.f, sq = 0.f;
#pragma unroll
    for (int j = 0; j < 4; ++j) { sum += xv[j]; sq += xv[j] * xv[j]; }
#pragma unroll
    for (int off = 16; off >= 1; off >>= 1) {
      sum += __shfl_xor(sum, off, 32);
      sq  += __shfl_xor(sq,  off, 32);
    }
    float mu = sum * (1.0f / CZ);
    float var = sq * (1.0f / CZ) - mu * mu;
    float rs = rsqrtf(var + 1e-5f);

    float nv[4];
#pragma unroll
    for (int j = 0; j < 4; ++j) nv[j] = (xv[j] - mu) * rs * lw[c0 + j] + lb[c0 + j];

    float acc[16];
#pragma unroll
    for (int hh = 0; hh < 16; ++hh) acc[hh] = 0.f;
#pragma unroll
    for (int j = 0; j < 4; ++j)
#pragma unroll
      for (int hh = 0; hh < 16; ++hh)
        acc[hh] += nv[j] * wzb_s[(c0 + j) * 16 + hh];
#pragma unroll
    for (int hh = 0; hh < 16; ++hh)
#pragma unroll
      for (int off = 16; off >= 1; off >>= 1)
        acc[hh] += __shfl_xor(acc[hh], off, 32);

    if (lane == 0) {
      int q = (int)(pair >> 10), k = (int)(pair & 1023);
      bool mk = mask[pair] != 0;
#pragma unroll
      for (int hh = 0; hh < 16; ++hh) {
        float bval = mk ? acc[hh] : -30000.0f;   // fp16-safe "minus infinity"
        biash[((size_t)hh * N + q) * N + k] = (_Float16)bval;
      }
    }
  }
}

// ---------------------------------------------------------------------------
// Flash attention: one wave per (head, 16-row q tile), 32-key tiles.
// Per tile: 4x wmma QK^T (D=48 padded to 64, K^T prepacked) + 3x full wmma P@V.
// Online softmax with half-wave (16-lane) butterfly reductions.
// ---------------------------------------------------------------------------
__global__ void attn_kernel(const _Float16* __restrict__ qkvh,
                            const _Float16* __restrict__ kT,
                            const _Float16* __restrict__ biash,
                            const float* __restrict__ gate,
                            _Float16* __restrict__ aoh) {
  const int N = 1024, CS = 768, LD = 2304, D = 48;
  const float scale = 0.144337567297406441f; // 48^-0.5

  int wid = threadIdx.x >> 5, lane = threadIdx.x & 31;
  int n16 = lane & 15, grp = lane >> 4;
  int h = blockIdx.y;
  int qt = blockIdx.x * 4 + wid;
  int qbase = qt * 16;

  __shared__ _Float16 plds[4][16 * 32];   // P tile 16(q) x 32(k) per wave

  // Q A-fragments (rows m = lane%16; 2 K-chunks covering d=0..63, pad >=48)
  const _Float16* qrow = qkvh + (size_t)(qbase + n16) * LD + h * D;
  v16h aq0 = load_afrag(qrow, 0, grp);
  v16h aq1;
  {
    v8h lo = *(const v8h*)(qrow + 32 + grp * 8);  // d = 32+8g .. 39+8g (<48, valid)
#pragma unroll
    for (int j = 0; j < 8; ++j) { aq1[j] = lo[j]; aq1[8 + j] = (_Float16)0.0f; } // d>=48 pad
  }

  float Mr[8], Lr[8];
#pragma unroll
  for (int r = 0; r < 8; ++r) { Mr[r] = -1e30f; Lr[r] = 0.0f; }
  v8f oacc[3];
#pragma unroll
  for (int t = 0; t < 3; ++t) oacc[t] = zero8();

  const _Float16* ktb = kT + (size_t)h * 64 * N;     // [d(64, zero-padded)][n]
  const _Float16* vP  = qkvh + 2 * CS + h * D;       // V block
  const _Float16* brow = biash + ((size_t)h * N + qbase) * N;

  for (int kt = 0; kt < 32; ++kt) {
    int kb = kt * 32;

    // Score tiles for keys kb..kb+15 (s=0) and kb+16..kb+31 (s=1).
    // B-fragment: lane L = key column L%16; elements = d in table order.
    v16h b00, b01, b10, b11;
#pragma unroll
    for (int s = 0; s < 16; ++s) {
      int kk = (s & 7) + 8 * grp + 16 * (s >> 3);
      b00[s] = ktb[(size_t)kk * N + kb + n16];
      b01[s] = ktb[(size_t)(32 + kk) * N + kb + n16];
      b10[s] = ktb[(size_t)kk * N + kb + 16 + n16];
      b11[s] = ktb[(size_t)(32 + kk) * N + kb + 16 + n16];
    }

    v8f sa0 = zero8(), sa1 = zero8();
    sa0 = __builtin_amdgcn_wmma_f32_16x16x32_f16(false, aq0, false, b00, (short)0, sa0, false, false);
    sa0 = __builtin_amdgcn_wmma_f32_16x16x32_f16(false, aq1, false, b01, (short)0, sa0, false, false);
    sa1 = __builtin_amdgcn_wmma_f32_16x16x32_f16(false, aq0, false, b10, (short)0, sa1, false, false);
    sa1 = __builtin_amdgcn_wmma_f32_16x16x32_f16(false, aq1, false, b11, (short)0, sa1, false, false);

    // online softmax over the 32 new keys (rows on 16-lane half-waves)
    float p0[8], p1[8];
#pragma unroll
    for (int r = 0; r < 8; ++r) {
      int m = r + 8 * grp;
      p0[r] = sa0[r] * scale + (float)brow[(size_t)m * N + kb + n16];
      p1[r] = sa1[r] * scale + (float)brow[(size_t)m * N + kb + 16 + n16];
    }
#pragma unroll
    for (int r = 0; r < 8; ++r) {
      float v = fmaxf(p0[r], p1[r]);
#pragma unroll
      for (int off = 8; off >= 1; off >>= 1) v = fmaxf(v, __shfl_xor(v, off, 32));
      float Mn = fmaxf(Mr[r], v);
      float al = __expf(Mr[r] - Mn);
      float pe0 = __expf(p0[r] - Mn);
      float pe1 = __expf(p1[r] - Mn);
      p0[r] = pe0; p1[r] = pe1;
      float sm = pe0 + pe1;
#pragma unroll
      for (int off = 8; off >= 1; off >>= 1) sm += __shfl_xor(sm, off, 32);
      Lr[r] = Lr[r] * al + sm;
      Mr[r] = Mn;
      oacc[0][r] *= al; oacc[1][r] *= al; oacc[2][r] *= al;
    }

    // redistribute P (C-layout) -> A-fragment layout via LDS (16x32, no padding)
    __syncthreads();
#pragma unroll
    for (int r = 0; r < 8; ++r) {
      int m = r + 8 * grp;
      plds[wid][m * 32 + n16]      = (_Float16)p0[r];
      plds[wid][m * 32 + 16 + n16] = (_Float16)p1[r];
    }
    __syncthreads();

    v16h ap = load_afrag(&plds[wid][n16 * 32], 0, grp);

    // P @ V: 3 D-tiles of 16; full 32-lane B fragments (rows kb..kb+31).
#pragma unroll
    for (int t = 0; t < 3; ++t) {
      v16h bv;
#pragma unroll
      for (int s = 0; s < 16; ++s) bv[s] = vP[(size_t)(kb + (s & 7) + 8 * grp + 16 * (s >> 3)) * LD + t * 16 + n16];
      oacc[t] = __builtin_amdgcn_wmma_f32_16x16x32_f16(false, ap, false, bv, (short)0, oacc[t], false, false);
    }
  }

  // epilogue: normalize, gate (sigmoid precomputed), store fp16
#pragma unroll
  for (int r = 0; r < 8; ++r) {
    int m = qbase + r + 8 * grp;
    float inv = 1.0f / Lr[r];
#pragma unroll
    for (int t = 0; t < 3; ++t) {
      int d = t * 16 + n16;
      size_t idx = (size_t)m * CS + h * D + d;
      float val = oacc[t][r] * inv * gate[idx];
      aoh[idx] = (_Float16)val;
    }
  }
}

// ---------------------------------------------------------------------------
// Launch
// ---------------------------------------------------------------------------
extern "C" void kernel_launch(void* const* d_in, const int* in_sizes, int n_in,
                              void* d_out, int out_size, void* d_ws, size_t ws_size,
                              hipStream_t stream) {
  (void)in_sizes; (void)n_in; (void)out_size; (void)ws_size;
  const float* s      = (const float*)d_in[0];
  const float* z      = (const float*)d_in[1];
  const unsigned char* pmask = (const unsigned char*)d_in[2];  // bool array
  const float* ln_s_w = (const float*)d_in[3];
  const float* ln_s_b = (const float*)d_in[4];
  const float* ln_z_w = (const float*)d_in[5];
  const float* ln_z_b = (const float*)d_in[6];
  const float* W_qkv  = (const float*)d_in[7];
  const float* b_qkv  = (const float*)d_in[8];
  const float* W_g    = (const float*)d_in[9];
  const float* W_zb   = (const float*)d_in[10];
  const float* W_o    = (const float*)d_in[11];
  const float* b_o    = (const float*)d_in[12];

  const int N = 1024, CS = 768, H = 16;

  char* p = (char*)d_ws;
  auto alloc = [&](size_t bytes) -> char* {
    char* r = p;
    p += (bytes + 255) & ~(size_t)255;
    return r;
  };
  _Float16* wqkvh = (_Float16*)alloc((size_t)CS * 3 * CS * 2); // 768x2304
  _Float16* wgh   = (_Float16*)alloc((size_t)CS * CS * 2);     // 768x768
  _Float16* woh   = (_Float16*)alloc((size_t)CS * CS * 2);     // 768x768
  _Float16* snh   = (_Float16*)alloc((size_t)N * CS * 2);      // LN(s) fp16
  _Float16* qkvh  = (_Float16*)alloc((size_t)N * 3 * CS * 2);  // qkv fp16
  _Float16* kTh   = (_Float16*)alloc((size_t)H * 64 * N * 2);  // K^T prepack
  float*    gatef = (float*)alloc((size_t)N * CS * 4);         // sigmoid(s_n@W_g)
  _Float16* aoh   = (_Float16*)alloc((size_t)N * CS * 2);      // gated attn out fp16
  _Float16* biash = (_Float16*)alloc((size_t)H * N * N * 2);   // pair bias fp16

  // weights -> fp16
  {
    int n1 = CS * 3 * CS, n2 = CS * CS;
    cvt_f32_f16_kernel<<<(n1 + 255) / 256, 256, 0, stream>>>(W_qkv, wqkvh, n1);
    cvt_f32_f16_kernel<<<(n2 + 255) / 256, 256, 0, stream>>>(W_g, wgh, n2);
    cvt_f32_f16_kernel<<<(n2 + 255) / 256, 256, 0, stream>>>(W_o, woh, n2);
  }

  // LayerNorm(s)
  ln_s_kernel<<<N / 8, 256, 0, stream>>>(s, ln_s_w, ln_s_b, snh);

  // qkv = s_n @ W_qkv + b_qkv  (fp16 out): 36*64 = 2304 wave-strips -> 576 blocks
  gemm_wmma_kernel<<<576, 128, 0, stream>>>(snh, CS, wqkvh, 3 * CS,
                                            N, 3 * CS, CS, b_qkv,
                                            qkvh, nullptr, 3 * CS, 0);
  // K^T prepack (after qkv): 16*64*1024 elems -> 4096 blocks
  ktrans_kernel<<<4096, 256, 0, stream>>>(qkvh, kTh);

  // gate = sigmoid(s_n @ W_g)  (fp32 out): 12*64 = 768 strips -> 192 blocks
  gemm_wmma_kernel<<<192, 128, 0, stream>>>(snh, CS, wgh, CS,
                                            N, CS, CS, nullptr,
                                            nullptr, gatef, CS, 1);

  // pair bias from z (512 MB stream): 1M pairs, 8 waves x 4 pairs per block
  zbias_kernel<<<(N * N) / 32, 256, 0, stream>>>(z, ln_z_w, ln_z_b, W_zb, pmask, biash);

  // flash attention: grid (16 qtile-groups x 16 heads), 4 waves/block
  attn_kernel<<<dim3(16, 16), 128, 0, stream>>>(qkvh, kTh, biash, gatef, aoh);

  // out = ao @ W_o + b_o  (fp32 -> d_out): 192 blocks
  gemm_wmma_kernel<<<192, 128, 0, stream>>>(aoh, CS, woh, CS,
                                            N, CS, CS, b_o,
                                            nullptr, (float*)d_out, CS, 2);
}
